// DeltaNetBlock_85083302133837
// MI455X (gfx1250) — compile-verified, matrix operand order checked by reading, not run on hardware
//
#include <hip/hip_runtime.h>
#include <hip/hip_bf16.h>
#include <math.h>

// ---------------- problem constants ----------------
#define Bq   2
#define Tq   1024
#define Dq   2048
#define Hq   32
#define DKq  48
#define DVq  96
#define KDq  1536   // H*DK
#define VDq  3072   // H*DV
#define CSq  4
#define INTERq 5632
#define Mrows (Bq*Tq)   // 2048

typedef __bf16 bf16x16 __attribute__((ext_vector_type(16)));
typedef float  floatx8 __attribute__((ext_vector_type(8)));
typedef unsigned int u32x4 __attribute__((ext_vector_type(4)));
typedef int i32x4 __attribute__((ext_vector_type(4)));
typedef int i32x8 __attribute__((ext_vector_type(8)));

__device__ __forceinline__ float silu_f(float x) { return x / (1.0f + __expf(-x)); }

#if __has_builtin(__builtin_amdgcn_tensor_load_to_lds)
#define HAVE_TDM 1
#else
#define HAVE_TDM 0
#endif

#if HAVE_TDM
// Issue a 2-D TDM tile load: global (bf16, row stride = stride_e elems) -> LDS.
// D# layout per CDNA5 ISA (group0: count/lds/global/type,
// group1: data_size=2B, tensor dims, tile dims, dim0 stride).
__device__ __forceinline__ void tdm_load_tile_bf16(unsigned lds_off,
                                                   const __bf16* gptr,
                                                   unsigned tensor_w, unsigned tensor_h,
                                                   unsigned stride_e,
                                                   unsigned tile_w, unsigned tile_h) {
  unsigned long long ga = (unsigned long long)(size_t)gptr;
  u32x4 g0;
  g0[0] = 1u;                                              // count=1, is_restore=0
  g0[1] = lds_off;                                         // lds_addr (bytes)
  g0[2] = (unsigned)(ga & 0xffffffffu);                    // global_addr[31:0]
  g0[3] = (unsigned)((ga >> 32) & 0x01ffffffu) | (2u << 30); // addr[56:32] | type=2
  i32x8 g1;
  g1[0] = (int)(1u << 16);                                 // data_size=1 -> 2 bytes
  g1[1] = (int)((tensor_w & 0xffffu) << 16);               // tensor_dim0[15:0]
  g1[2] = (int)(((tensor_w >> 16) & 0xffffu) | ((tensor_h & 0xffffu) << 16));
  g1[3] = (int)(((tensor_h >> 16) & 0xffffu) | (tile_w << 16)); // tile_dim0
  g1[4] = (int)(tile_h & 0xffffu);                         // tile_dim1 (tile_dim2=0)
  g1[5] = (int)stride_e;                                   // tensor_dim0_stride[31:0]
  g1[6] = 0;
  g1[7] = 0;
  i32x4 z4; z4[0] = 0; z4[1] = 0; z4[2] = 0; z4[3] = 0;
#if defined(__clang_major__) && __clang_major__ >= 23
  i32x8 z8; z8[0]=0; z8[1]=0; z8[2]=0; z8[3]=0; z8[4]=0; z8[5]=0; z8[6]=0; z8[7]=0;
  __builtin_amdgcn_tensor_load_to_lds(g0, g1, z4, z4, z8, 0);
#else
  __builtin_amdgcn_tensor_load_to_lds(g0, g1, z4, z4, 0);
#endif
}
#endif

// ---------------- WMMA GEMM: C[M,N] = A[M,K] * Bt[N,K]^T, bf16 in, f32 out ----
// A is [M,K] bf16 row-major; Bt is [N,K] bf16 row-major (pre-transposed weight).
#define BM 128
#define BN 128
#define BK 32
#define TILE_E (128 * BK)          // elements per LDS tile
#define TILE_B (TILE_E * 2)        // bytes per LDS tile (8KB)

__global__ __launch_bounds__(256)
void gemm_wmma_bf16(const __bf16* __restrict__ A, const __bf16* __restrict__ Bt,
                    float* __restrict__ C, int M, int N, int K) {
  // [ A buf0 | A buf1 | B buf0 | B buf1 ], each 128x32 bf16 (8KB)
  __shared__ __attribute__((aligned(16))) __bf16 smem[4 * TILE_E];

  const int tid  = threadIdx.x;
  const int wave = tid >> 5;
  const int lane = tid & 31;
  const int wm   = wave >> 2;           // 0..1  -> 64 rows each
  const int wn   = wave & 3;            // 0..3  -> 32 cols each
  const int tileM = blockIdx.y * BM;
  const int tileN = blockIdx.x * BN;
  const int wave0 = (__builtin_amdgcn_readfirstlane(wave) == 0);

  // Escape the LDS object (input-only, so accesses keep addrspace(3) and
  // lower to ds_load): the TDM engine writes smem behind the compiler's back.
  asm volatile("" :: "s"(smem) : "memory");

  floatx8 acc[4][2];
  #pragma unroll
  for (int i = 0; i < 4; ++i)
    #pragma unroll
    for (int j = 0; j < 2; ++j)
      #pragma unroll
      for (int e = 0; e < 8; ++e) acc[i][j][e] = 0.0f;

  const int nk = K / BK;

#if HAVE_TDM
  // ---- prologue: DMA tiles for k-step 0 into buffer 0 ----
  if (wave0) {
    tdm_load_tile_bf16(0u,          A  + (size_t)tileM * K, (unsigned)K, 128u,
                       (unsigned)K, BK, 128u);
    tdm_load_tile_bf16(2u * TILE_B, Bt + (size_t)tileN * K, (unsigned)K, 128u,
                       (unsigned)K, BK, 128u);
  }
#endif

  for (int j = 0; j < nk; ++j) {
    const int buf = j & 1;
#if HAVE_TDM
    const int more = (j + 1 < nk);
    if (wave0) {
      if (more) {
        const int k1 = (j + 1) * BK;
        tdm_load_tile_bf16((unsigned)((buf ^ 1) * TILE_B),
                           A  + (size_t)tileM * K + k1, (unsigned)K, 128u,
                           (unsigned)K, BK, 128u);
        tdm_load_tile_bf16((unsigned)(2 * TILE_B + (buf ^ 1) * TILE_B),
                           Bt + (size_t)tileN * K + k1, (unsigned)K, 128u,
                           (unsigned)K, BK, 128u);
        __builtin_amdgcn_s_wait_tensorcnt(2);   // keep the 2 new DMAs in flight
      } else {
        __builtin_amdgcn_s_wait_tensorcnt(0);   // drain for the last tile
      }
    }
#else
    // fallback staging: vectorized global loads, direct bf16 copy
    {
      const int k0 = j * BK;
      #pragma unroll
      for (int it = 0; it < 2; ++it) {
        int idx = tid + it * 256;             // 0..511, one uint4 (8 bf16) each
        int r   = idx >> 2;                   // row 0..127
        int c8  = (idx & 3) * 8;              // col 0,8,16,24
        *(uint4*)&smem[buf * TILE_E + r * BK + c8] =
            *(const uint4*)(A + (size_t)(tileM + r) * K + k0 + c8);
        *(uint4*)&smem[(2 + buf) * TILE_E + r * BK + c8] =
            *(const uint4*)(Bt + (size_t)(tileN + r) * K + k0 + c8);
      }
    }
#endif
    // Per-iteration clobber: LDS contents changed (TDM DMA / other waves).
    asm volatile("" ::: "memory");
    __syncthreads();

    const __bf16* sA = &smem[buf * TILE_E];
    const __bf16* sB = &smem[(2 + buf) * TILE_E];
    const int halfk = (lane >> 4) * 8;        // K base per lane half (0 or 8)

    bf16x16 bfrag[2];
    #pragma unroll
    for (int jj = 0; jj < 2; ++jj) {
      int col = wn * 32 + jj * 16 + (lane & 15);
      uint4* p = (uint4*)&bfrag[jj];
      p[0] = *(const uint4*)&sB[col * BK + halfk];
      p[1] = *(const uint4*)&sB[col * BK + 16 + halfk];
    }
    #pragma unroll
    for (int i = 0; i < 4; ++i) {
      int row = wm * 64 + i * 16 + (lane & 15);
      bf16x16 afrag;
      uint4* p = (uint4*)&afrag;
      p[0] = *(const uint4*)&sA[row * BK + halfk];
      p[1] = *(const uint4*)&sA[row * BK + 16 + halfk];
      #pragma unroll
      for (int jj = 0; jj < 2; ++jj)
        acc[i][jj] = __builtin_amdgcn_wmma_f32_16x16x32_bf16(
            false, afrag, false, bfrag[jj], (short)0, acc[i][jj], false, false);
    }
    __syncthreads();
  }

  #pragma unroll
  for (int i = 0; i < 4; ++i)
    #pragma unroll
    for (int jj = 0; jj < 2; ++jj) {
      int row0 = tileM + wm * 64 + i * 16 + (lane >> 4) * 8;
      int col  = tileN + wn * 32 + jj * 16 + (lane & 15);
      #pragma unroll
      for (int e = 0; e < 8; ++e)
        C[(size_t)(row0 + e) * N + col] = acc[i][jj][e];
    }
}

// ---------------- RMSNorm -> bf16 ----------------
__global__ __launch_bounds__(256)
void rmsnorm_bf16_kernel(const float* __restrict__ x, const float* __restrict__ w,
                         __bf16* __restrict__ out, int D) {
  int row = blockIdx.x;
  const float* xr = x + (size_t)row * D;
  __shared__ float red[256];
  float s = 0.0f;
  for (int i = threadIdx.x; i < D; i += 256) { float v = xr[i]; s += v * v; }
  red[threadIdx.x] = s; __syncthreads();
  for (int off = 128; off > 0; off >>= 1) {
    if (threadIdx.x < off) red[threadIdx.x] += red[threadIdx.x + off];
    __syncthreads();
  }
  float r = rsqrtf(red[0] / (float)D + 1e-5f);
  for (int i = threadIdx.x; i < D; i += 256)
    out[(size_t)row * D + i] = (__bf16)(xr[i] * r * w[i]);
}

// ---------------- W[K,N] f32 -> Wt[N,K] bf16 ----------------
__global__ __launch_bounds__(256)
void transpose_convert_kernel(const float* __restrict__ W, __bf16* __restrict__ Wt,
                              int K, int N, int total) {
  int idx = blockIdx.x * 256 + threadIdx.x;
  if (idx >= total) return;
  int k = idx % K;
  int n = idx / K;
  Wt[(size_t)n * K + k] = (__bf16)W[(size_t)k * N + n];
}

// ---------------- causal depthwise conv (K=4) + silu ----------------
__global__ __launch_bounds__(256)
void conv_silu_kernel(const float* __restrict__ xin, const float* __restrict__ w,
                      float* __restrict__ out, int C, int total) {
  int idx = blockIdx.x * 256 + threadIdx.x;
  if (idx >= total) return;
  int c = idx % C;
  int t = (idx / C) % Tq;
  int b = idx / (C * Tq);
  float s = 0.0f;
  #pragma unroll
  for (int k = 0; k < CSq; ++k) {
    int tt = t - (CSq - 1) + k;
    if (tt >= 0) s += xin[((size_t)b * Tq + tt) * C + c] * w[c * CSq + k];
  }
  out[idx] = silu_f(s);
}

// ---------------- per-head l2norm (in place) ----------------
__global__ __launch_bounds__(256)
void l2norm_kernel(float* __restrict__ qk, int total_rh) {
  int idx = blockIdx.x * 256 + threadIdx.x;
  if (idx >= total_rh) return;
  int hh = idx % Hq, row = idx / Hq;
  float* p = qk + (size_t)row * KDq + hh * DKq;
  float s = 0.0f;
  #pragma unroll
  for (int i = 0; i < DKq; ++i) s += p[i] * p[i];
  float r = rsqrtf(s + 1e-6f);
  #pragma unroll
  for (int i = 0; i < DKq; ++i) p[i] *= r;
}

// ---------------- g / beta ----------------
__global__ __launch_bounds__(256)
void g_beta_kernel(const __bf16* __restrict__ h, const float* __restrict__ Wa,
                   const float* __restrict__ Wb, const float* __restrict__ dt_bias,
                   const float* __restrict__ A_log, float* __restrict__ g,
                   float* __restrict__ beta, int total_rh) {
  int idx = blockIdx.x * 256 + threadIdx.x;
  if (idx >= total_rh) return;
  int hh = idx % Hq, row = idx / Hq;
  const __bf16* hr = h + (size_t)row * Dq;
  float a = 0.0f, bv = 0.0f;
  for (int d = 0; d < Dq; ++d) {
    float hv = (float)hr[d];
    a  += hv * Wa[(size_t)d * Hq + hh];
    bv += hv * Wb[(size_t)d * Hq + hh];
  }
  float xg = a + dt_bias[hh];
  float sp = (xg > 20.0f) ? xg : log1pf(__expf(xg));
  g[idx]    = -__expf(A_log[hh]) * sp;
  beta[idx] = 1.0f / (1.0f + __expf(-bv));
}

// ---------------- gated delta rule: one block per (b,h), 96 threads ----------
__global__ __launch_bounds__(96)
void delta_rule_kernel(const float* __restrict__ qn, const float* __restrict__ kn,
                       const float* __restrict__ vc, const float* __restrict__ g,
                       const float* __restrict__ beta, float* __restrict__ od) {
  const int bh = blockIdx.x;
  const int b  = bh >> 5;
  const int hh = bh & 31;
  const int j  = threadIdx.x;
  __shared__ float ks[DKq], qs[DKq], gb[2];
  float S[DKq];
  #pragma unroll
  for (int i = 0; i < DKq; ++i) S[i] = 0.0f;
  const float scale = 0.1443375673f;   // 48^-0.5

  for (int t = 0; t < Tq; ++t) {
    size_t row = (size_t)b * Tq + t;
    if (j < DKq) {
      ks[j] = kn[row * KDq + hh * DKq + j];
      qs[j] = qn[row * KDq + hh * DKq + j];
    } else if (j == DKq) {
      gb[0] = g[row * Hq + hh];
      gb[1] = beta[row * Hq + hh];
    }
    __syncthreads();
    float eg = __expf(gb[0]);
    float bt = gb[1];
    float vt = vc[row * VDq + hh * DVq + j];
    float mem = 0.0f;
    #pragma unroll
    for (int i = 0; i < DKq; ++i) { S[i] *= eg; mem += S[i] * ks[i]; }
    float delta = (vt - mem) * bt;
    float o = 0.0f;
    #pragma unroll
    for (int i = 0; i < DKq; ++i) { S[i] += ks[i] * delta; o += qs[i] * S[i]; }
    od[row * VDq + hh * DVq + j] = o * scale;
    __syncthreads();
  }
}

// ---------------- gated per-head RMSNorm * silu(gate) -> bf16 ----------------
__global__ __launch_bounds__(256)
void gated_norm_kernel(const float* __restrict__ od, const float* __restrict__ gw,
                       const float* __restrict__ gate, __bf16* __restrict__ og,
                       int total_rh) {
  int idx = blockIdx.x * 256 + threadIdx.x;
  if (idx >= total_rh) return;
  int hh = idx % Hq, row = idx / Hq;
  const float* o  = od   + (size_t)row * VDq + hh * DVq;
  const float* gt = gate + (size_t)row * VDq + hh * DVq;
  __bf16* out = og + (size_t)row * VDq + hh * DVq;
  float s = 0.0f;
  #pragma unroll
  for (int i = 0; i < DVq; ++i) s += o[i] * o[i];
  float r = rsqrtf(s / (float)DVq + 1e-5f);
  #pragma unroll
  for (int i = 0; i < DVq; ++i)
    out[i] = (__bf16)(o[i] * r * gw[i] * silu_f(gt[i]));
}

// ---------------- elementwise ----------------
__global__ __launch_bounds__(256)
void add_kernel(const float* __restrict__ a, const float* __restrict__ b,
                float* __restrict__ out, int n) {
  int i = blockIdx.x * 256 + threadIdx.x;
  if (i < n) out[i] = a[i] + b[i];
}

__global__ __launch_bounds__(256)
void swiglu_bf16_kernel(const float* __restrict__ u1, const float* __restrict__ u3,
                        __bf16* __restrict__ out, int n) {
  int i = blockIdx.x * 256 + threadIdx.x;
  if (i < n) out[i] = (__bf16)(silu_f(u1[i]) * u3[i]);
}

// ---------------- host launcher ----------------
static inline dim3 gemm_grid(int M, int N) { return dim3(N / BN, M / BM); }

extern "C" void kernel_launch(void* const* d_in, const int* in_sizes, int n_in,
                              void* d_out, int out_size, void* d_ws, size_t ws_size,
                              hipStream_t stream) {
  const float* x       = (const float*)d_in[0];
  const float* rms1_w  = (const float*)d_in[1];
  const float* rms2_w  = (const float*)d_in[2];
  const float* Wq      = (const float*)d_in[3];
  const float* Wk      = (const float*)d_in[4];
  const float* Wv      = (const float*)d_in[5];
  const float* Wa      = (const float*)d_in[6];
  const float* Wb      = (const float*)d_in[7];
  const float* Wg      = (const float*)d_in[8];
  const float* dt_bias = (const float*)d_in[9];
  const float* A_log   = (const float*)d_in[10];
  const float* cqw     = (const float*)d_in[11];
  const float* ckw     = (const float*)d_in[12];
  const float* cvw     = (const float*)d_in[13];
  const float* gnorm_w = (const float*)d_in[14];
  const float* Wo      = (const float*)d_in[15];
  const float* W1      = (const float*)d_in[16];
  const float* W3      = (const float*)d_in[17];
  const float* W2      = (const float*)d_in[18];
  float* out = (float*)d_out;

  char* ws = (char*)d_ws;
  const size_t MB = 1ull << 20;
  __bf16* Wt    = (__bf16*)(ws + 0 * MB);      // 22MB  [0,22)  transposed-weight scratch
  __bf16* h_bf  = (__bf16*)(ws + 22 * MB);     // 8MB   [22,30)  (later: ff_bf)
  float*  qlin  = (float*)(ws + 30 * MB);      // 12MB  [30,42)
  float*  klin  = (float*)(ws + 42 * MB);      // 12MB  [42,54)
  float*  vlin  = (float*)(ws + 54 * MB);      // 24MB  [54,78)
  float*  gate  = (float*)(ws + 78 * MB);      // 24MB  [78,102)
  float*  gbuf  = (float*)(ws + 102 * MB);     // 256KB
  float*  bbuf  = (float*)(ws + 102 * MB + 256 * 1024);
  float*  qn    = (float*)(ws + 103 * MB);     // 12MB  [103,115)
  float*  kn    = (float*)(ws + 115 * MB);     // 12MB  [115,127)
  float*  vc    = (float*)(ws + 127 * MB);     // 24MB  [127,151)
  float*  odel  = (float*)(ws + 151 * MB);     // 24MB  [151,175)
  __bf16* og_bf = (__bf16*)(ws + 30 * MB);     // 12MB  reuse [30,42)
  float*  attn  = (float*)(ws + 42 * MB);      // 16MB  reuse [42,58)
  float*  h2    = (float*)(ws + 58 * MB);      // 16MB  [58,74)
  __bf16* ff_bf = (__bf16*)(ws + 22 * MB);     // 8MB   reuse [22,30)
  float*  w1o   = (float*)(ws + 103 * MB);     // 44MB  reuse [103,147)
  float*  w3o   = (float*)(ws + 147 * MB);     // 44MB  [147,191)
  __bf16* u_bf  = (__bf16*)(ws + 30 * MB);     // 22MB  reuse [30,52)
  float*  mlp   = (float*)(ws + 74 * MB);      // 16MB  reuse [74,90)
  (void)ws_size; (void)in_sizes; (void)n_in; (void)out_size;

  #define TRANSPOSE(W_, K_, N_) \
    transpose_convert_kernel<<<((K_) * (N_) + 255) / 256, 256, 0, stream>>>(W_, Wt, K_, N_, (K_) * (N_))

  // 1) h = rmsnorm(x) -> bf16
  rmsnorm_bf16_kernel<<<Mrows, 256, 0, stream>>>(x, rms1_w, h_bf, Dq);

  // 2) linears (TDM-staged WMMA)
  TRANSPOSE(Wq, Dq, KDq);
  gemm_wmma_bf16<<<gemm_grid(Mrows, KDq), 256, 0, stream>>>(h_bf, Wt, qlin, Mrows, KDq, Dq);
  TRANSPOSE(Wk, Dq, KDq);
  gemm_wmma_bf16<<<gemm_grid(Mrows, KDq), 256, 0, stream>>>(h_bf, Wt, klin, Mrows, KDq, Dq);
  TRANSPOSE(Wv, Dq, VDq);
  gemm_wmma_bf16<<<gemm_grid(Mrows, VDq), 256, 0, stream>>>(h_bf, Wt, vlin, Mrows, VDq, Dq);
  TRANSPOSE(Wg, Dq, VDq);
  gemm_wmma_bf16<<<gemm_grid(Mrows, VDq), 256, 0, stream>>>(h_bf, Wt, gate, Mrows, VDq, Dq);
  {
    int total = Mrows * Hq;
    g_beta_kernel<<<(total + 255) / 256, 256, 0, stream>>>(h_bf, Wa, Wb, dt_bias, A_log,
                                                           gbuf, bbuf, total);
  }

  // 3) conv + silu
  {
    int tq = Mrows * KDq, tv = Mrows * VDq;
    conv_silu_kernel<<<(tq + 255) / 256, 256, 0, stream>>>(qlin, cqw, qn, KDq, tq);
    conv_silu_kernel<<<(tq + 255) / 256, 256, 0, stream>>>(klin, ckw, kn, KDq, tq);
    conv_silu_kernel<<<(tv + 255) / 256, 256, 0, stream>>>(vlin, cvw, vc, VDq, tv);
  }

  // 4) l2norm q,k
  {
    int total = Mrows * Hq;
    l2norm_kernel<<<(total + 255) / 256, 256, 0, stream>>>(qn, total);
    l2norm_kernel<<<(total + 255) / 256, 256, 0, stream>>>(kn, total);
  }

  // 5) gated delta rule scan
  delta_rule_kernel<<<Bq * Hq, DVq, 0, stream>>>(qn, kn, vc, gbuf, bbuf, odel);

  // 6) gated per-head RMSNorm * silu(gate) -> bf16
  {
    int total = Mrows * Hq;
    gated_norm_kernel<<<(total + 255) / 256, 256, 0, stream>>>(odel, gnorm_w, gate, og_bf, total);
  }

  // 7) out proj + residual
  TRANSPOSE(Wo, VDq, Dq);
  gemm_wmma_bf16<<<gemm_grid(Mrows, Dq), 256, 0, stream>>>(og_bf, Wt, attn, Mrows, Dq, VDq);
  add_kernel<<<(Mrows * Dq + 255) / 256, 256, 0, stream>>>(x, attn, h2, Mrows * Dq);

  // 8) MLP
  rmsnorm_bf16_kernel<<<Mrows, 256, 0, stream>>>(h2, rms2_w, ff_bf, Dq);
  TRANSPOSE(W1, Dq, INTERq);
  gemm_wmma_bf16<<<gemm_grid(Mrows, INTERq), 256, 0, stream>>>(ff_bf, Wt, w1o, Mrows, INTERq, Dq);
  TRANSPOSE(W3, Dq, INTERq);
  gemm_wmma_bf16<<<gemm_grid(Mrows, INTERq), 256, 0, stream>>>(ff_bf, Wt, w3o, Mrows, INTERq, Dq);
  swiglu_bf16_kernel<<<(Mrows * INTERq + 255) / 256, 256, 0, stream>>>(w1o, w3o, u_bf, Mrows * INTERq);
  TRANSPOSE(W2, INTERq, Dq);
  gemm_wmma_bf16<<<gemm_grid(Mrows, Dq), 256, 0, stream>>>(u_bf, Wt, mlp, Mrows, Dq, INTERq);

  // 9) final residual
  add_kernel<<<(Mrows * Dq + 255) / 256, 256, 0, stream>>>(h2, mlp, out, Mrows * Dq);
  #undef TRANSPOSE
}
